// WaveNetModel_62723702390788
// MI455X (gfx1250) — matrix operand order repaired and checked by reading, not compiled, not run on hardware
//
#include <hip/hip_runtime.h>
#include <hip/hip_bf16.h>
#include <math.h>

typedef __attribute__((ext_vector_type(16))) _Float16 v16h;
typedef __attribute__((ext_vector_type(8)))  _Float16 v8h;
typedef __attribute__((ext_vector_type(8)))  float    v8f;
typedef unsigned int u32;
typedef __attribute__((ext_vector_type(4))) u32 v4u;
typedef __attribute__((ext_vector_type(8))) int v8i;
typedef __attribute__((ext_vector_type(4))) int v4i;

// ---------------------------------------------------------------------------
// WMMA fragment helpers (wave32, V_WMMA_F32_16X16X32_F16)
// A (16x32 f16): two contiguous 8xf16 runs: [k0+half*8,+8) and [k0+16+half*8,+8)
// B (32x16 f16): same k order as A when X is [col][K]
// C/D: lane n = lane&15, vgpr v -> M = 8*(lane>>4)+v
// LDS row strides padded by +8 halves (16B-aligned rows, conflict-free b128s).
// ---------------------------------------------------------------------------
__device__ __forceinline__ v16h concat16(v8h lo, v8h hi) {
  return __builtin_shufflevector(lo, hi, 0, 1, 2, 3, 4, 5, 6, 7, 8, 9, 10, 11,
                                 12, 13, 14, 15);
}

__device__ __forceinline__ v16h load_a(const _Float16* W, int ldw, int m0, int k0) {
  int lane = threadIdx.x & 31;
  int half = lane >> 4;
  const _Float16* row = W + (m0 + (lane & 15)) * ldw + k0 + half * 8;
  v8h lo = *(const v8h*)row;
  v8h hi = *(const v8h*)(row + 16);
  return concat16(lo, hi);
}

__device__ __forceinline__ v16h load_b(const _Float16* X, int ldx, int c0, int k0) {
  int lane = threadIdx.x & 31;
  int half = lane >> 4;
  const _Float16* row = X + (c0 + (lane & 15)) * ldx + k0 + half * 8;
  v8h lo = *(const v8h*)row;
  v8h hi = *(const v8h*)(row + 16);
  return concat16(lo, hi);
}

__device__ __forceinline__ v8f wmma32(v16h a, v16h b, v8f c) {
  return __builtin_amdgcn_wmma_f32_16x16x32_f16(false, a, false, b, (short)0, c,
                                                false, false);
}

#define LOG2E 1.44269504f
__device__ __forceinline__ float fsigmoid(float x) {
  return __builtin_amdgcn_rcpf(1.f + __builtin_amdgcn_exp2f(-x * LOG2E));
}
__device__ __forceinline__ float ftanh(float x) {
  return 2.f * __builtin_amdgcn_rcpf(1.f + __builtin_amdgcn_exp2f(-2.f * LOG2E * x)) - 1.f;
}

// ---------------------------------------------------------------------------
// Tensor Data Mover: 2D f16 tile (tile0 cols x tile1 rows, row stride = stride0
// elements) global -> LDS.  LDS pad: +4 DWORDs after every 16 DWORDs => row
// stride 40 halves, matching the padded fragment layout.
// (clang-23 toolchain: 6-arg builtin form)
// ---------------------------------------------------------------------------
__device__ __forceinline__ void tdm_load_2d_f16(u32 lds_off, const _Float16* g,
                                                u32 dim0, u32 dim1, u32 stride0,
                                                u32 tile0, u32 tile1) {
  unsigned long long ga = (unsigned long long)(uintptr_t)g;
  v4u g0;
  g0[0] = 1u;                                  // count=1 user descriptor
  g0[1] = lds_off;                             // LDS byte address
  g0[2] = (u32)ga;                             // global addr lo
  g0[3] = (u32)(ga >> 32) | (2u << 30);        // global addr hi | type=2
  v8i g1;
  g1[0] = (int)((1u << 16) | (1u << 20) | (3u << 22) | (3u << 25));
  // data_size=1 (2B) | pad_enable | pad_interval=3 (16 dw) | pad_amount=3 (4 dw)
  g1[1] = (int)((dim0 & 0xFFFFu) << 16);
  g1[2] = (int)((dim0 >> 16) | ((dim1 & 0xFFFFu) << 16));
  g1[3] = (int)((dim1 >> 16) | (tile0 << 16));
  g1[4] = (int)tile1;
  g1[5] = (int)stride0;
  g1[6] = 0;
  g1[7] = 0;
  v4i z4 = {0, 0, 0, 0};
  v8i z8 = {0, 0, 0, 0, 0, 0, 0, 0};
  __builtin_amdgcn_tensor_load_to_lds(g0, g1, z4, z4, z8, 0);
}

// ---------------------------------------------------------------------------
// weight pre-conversion (once per launch): f32 -> f16
// ---------------------------------------------------------------------------
__global__ void k_cvt(const float* __restrict__ in, _Float16* __restrict__ out,
                      int n) {
  int idx = blockIdx.x * blockDim.x + threadIdx.x;
  if (idx < n) out[idx] = (_Float16)in[idx];
}

// filter/gate (6,32,32,2) -> (6,32,64) with kk = k*32 + i
__global__ void k_cvt_fg(const float* __restrict__ Wf, const float* __restrict__ Wg,
                         _Float16* __restrict__ of, _Float16* __restrict__ og,
                         int n) {
  int idx = blockIdx.x * blockDim.x + threadIdx.x;
  if (idx >= n) return;
  int kk = idx & 63;
  int m = (idx >> 6) & 31;
  int l = idx >> 11;
  int i = kk & 31, k = kk >> 5;
  int src = l * 2048 + (m * 32 + i) * 2 + k;
  of[idx] = (_Float16)Wf[src];
  og[idx] = (_Float16)Wg[src];
}

// ---------------------------------------------------------------------------
// start conv1x1: h[n,o,t] = sum_i w[o,i] * x[n,i,t]   (K=5, scalar)
// ---------------------------------------------------------------------------
__global__ void k_start(const float* __restrict__ x, const float* __restrict__ w,
                        float* __restrict__ h, int total) {
  int idx = blockIdx.x * blockDim.x + threadIdx.x;
  if (idx >= total) return;
  int t = idx % 1280;
  int o = (idx / 1280) & 31;
  int n = idx / (1280 * 32);
  const float* xr = x + n * 5 * 1280 + t;
  float acc = 0.f;
#pragma unroll
  for (int i = 0; i < 5; ++i) acc += w[o * 5 + i] * xr[i * 1280];
  h[idx] = acc;
}

// ---------------------------------------------------------------------------
// dilate (fold, f>=2): out[(k*Nin+j), c, i] = in_pad[j, c, i*f + k]
// ---------------------------------------------------------------------------
__global__ void k_fold(const float* __restrict__ in, float* __restrict__ out,
                       int Nin, int Lin, int f, int Lout, int pad, int total) {
  int idx = blockIdx.x * blockDim.x + threadIdx.x;
  if (idx >= total) return;
  int ip = idx % Lout;
  int c = (idx / Lout) & 31;
  int n2 = idx / (Lout * 32);
  int k = n2 / Nin, j = n2 - k * Nin;
  int t = ip * f + k - pad;
  out[idx] = (t < 0 || t >= Lin) ? 0.f : in[(j * 32 + c) * Lin + t];
}

// ---------------------------------------------------------------------------
// dilate (unfold, f = 1/g): out[j, c, i*g+k] = in[(k*Nout + j), c, i]
// ---------------------------------------------------------------------------
__global__ void k_unfold(const float* __restrict__ in, float* __restrict__ out,
                         int Nout, int Lin, int g, int total) {
  int idx = blockIdx.x * blockDim.x + threadIdx.x;
  if (idx >= total) return;
  int Lout = Lin * g;
  int tt = idx % Lout;
  int c = (idx / Lout) & 31;
  int jp = idx / (Lout * 32);
  int i = tt / g, k = tt - i * g;
  out[idx] = in[((k * Nout + jp) * 32 + c) * Lin + i];
}

// ---------------------------------------------------------------------------
// fused dilated gate layer: H = tanh(Wf (*) R) * sigmoid(Wg (*) R)
// ---------------------------------------------------------------------------
__global__ void __launch_bounds__(256)
k_conv2gate(const float* __restrict__ R, const _Float16* __restrict__ Wfh,
            const _Float16* __restrict__ Wgh, float* __restrict__ H, int Lr,
            int P) {
  __shared__ __align__(16) _Float16 wf[32 * 72];
  __shared__ __align__(16) _Float16 wg[32 * 72];
  __shared__ __align__(16) _Float16 xs[64 * 72];
  int Lo = Lr - 1;
  int tid = threadIdx.x;
  {                                         // 256 chunks of 8 halves: 1/thread
    int m = tid >> 3, q = tid & 7;
    *(v8h*)(wf + m * 72 + q * 8) = *(const v8h*)(Wfh + m * 64 + q * 8);
    *(v8h*)(wg + m * 72 + q * 8) = *(const v8h*)(Wgh + m * 64 + q * 8);
  }
  int p0 = blockIdx.x * 64;
  for (int e = tid; e < 4096; e += 256) {
    int c = e & 63, kk = e >> 6;
    int p = p0 + c;
    float v = 0.f;
    if (p < P) {
      int n = p / Lo, t = p - n * Lo;
      int i = kk & 31, k = kk >> 5;
      v = R[((size_t)n * 32 + i) * Lr + t + k];
    }
    xs[c * 72 + kk] = (_Float16)v;
  }
  __syncthreads();
  int w = tid >> 5;
  int m0 = (w & 1) * 16, c0 = (w >> 1) * 16;
  v8f accf = {}; v8f accg = {};
#pragma unroll
  for (int kb = 0; kb < 64; kb += 32) {
    v16h b = load_b(xs, 72, c0, kb);
    accf = wmma32(load_a(wf, 72, m0, kb), b, accf);
    accg = wmma32(load_a(wg, 72, m0, kb), b, accg);
  }
  int lane = tid & 31, half = lane >> 4, nn = lane & 15;
  int p = p0 + c0 + nn;
  if (p < P) {
    int n = p / Lo, t = p - n * Lo;
#pragma unroll
    for (int v = 0; v < 8; ++v) {
      int m = m0 + half * 8 + v;
      H[((size_t)n * 32 + m) * Lo + t] = ftanh(accf[v]) * fsigmoid(accg[v]);
    }
  }
}

// ---------------------------------------------------------------------------
// generic conv1x1 (WMMA): out[n,m,t(+off)] (=|+=) act( W[MxK] @ in[n,:,t] )
// flags: 1 = relu on input, 2 = relu on output, 4 = accumulate into out
// ---------------------------------------------------------------------------
__device__ __forceinline__ void stage_x(_Float16* xl, int ldx,
                                        const float* __restrict__ in, int p0,
                                        int P, int K, int KC, int kb0, int Lt,
                                        int inL, int in_t0, int relu_in) {
  for (int e = threadIdx.x; e < KC * 64; e += 256) {
    int c = e & 63, k = e >> 6;
    int p = p0 + c;
    float v = 0.f;
    if (p < P) {
      int n = p / Lt, t = p - n * Lt;
      v = in[((size_t)n * K + (kb0 + k)) * inL + t + in_t0];
      if (relu_in) v = fmaxf(v, 0.f);
    }
    xl[c * ldx + k] = (_Float16)v;
  }
}

__device__ __forceinline__ void store_tile(v8f acc, int tile, int p0, int P, int M,
                                           int Lt, int outL, int out_t0, int addL,
                                           int add_t0, const float* bias,
                                           const float* add, float* out, int flags) {
  int lane = threadIdx.x & 31, half = lane >> 4, nn = lane & 15;
  int m0 = (tile >> 2) * 16, c0 = (tile & 3) * 16;
  int p = p0 + c0 + nn;
  if (p >= P) return;
  int n = p / Lt, t = p - n * Lt;
#pragma unroll
  for (int v = 0; v < 8; ++v) {
    int m = m0 + half * 8 + v;
    float val = acc[v];
    if (bias) val += bias[m];
    if (add)  val += add[((size_t)n * M + m) * addL + t + add_t0];
    float* op = out + ((size_t)n * M + m) * outL + t + out_t0;
    if (flags & 4) val += *op;
    if (flags & 2) val = fmaxf(val, 0.f);
    *op = val;
  }
}

__global__ void __launch_bounds__(256)
k_conv1x1(const float* __restrict__ in, const _Float16* __restrict__ W,
          const float* __restrict__ bias, const float* __restrict__ add,
          float* __restrict__ out, int M, int K, int Lt, int inL, int in_t0,
          int outL, int out_t0, int addL, int add_t0, int P, int flags) {
  __shared__ __align__(16) _Float16 wl[10240];   // max(256*40, 128*72)
  __shared__ __align__(16) _Float16 xl[64 * 72];
  int tid = threadIdx.x;
  int p0 = blockIdx.x * 64;
  int w = tid >> 5;
  int ntiles = (M >> 4) << 2;

  if (K <= 64) {                             // Path 1 (skip 256x32, res 32x32)
    int lda = K + 8;
    int cs = (K == 64) ? 3 : 2;              // v8h chunks per row: 8 or 4
    int ck = K >> 3;
    for (int e = tid; e < M * ck; e += 256) {
      int m = e >> cs, q = e & (ck - 1);
      *(v8h*)(wl + m * lda + q * 8) = *(const v8h*)(W + m * K + q * 8);
    }
    stage_x(xl, lda, in, p0, P, K, K, 0, Lt, inL, in_t0, flags & 1);
    __syncthreads();
    for (int tile = w; tile < ntiles; tile += 8) {
      int m0 = (tile >> 2) * 16, c0 = (tile & 3) * 16;
      v8f acc = {};
      for (int kb = 0; kb < K; kb += 32)
        acc = wmma32(load_a(wl, lda, m0, kb), load_b(xl, lda, c0, kb), acc);
      store_tile(acc, tile, p0, P, M, Lt, outL, out_t0, addL, add_t0, bias, add,
                 out, flags);
    }
  } else {                                   // Path 2 (end1 128x256, end2 32x128)
    int tpw = ntiles >> 3;                   // tiles per wave: 1 or 4
    v8f a0 = {}, a1 = {}, a2 = {}, a3 = {};
    for (int kb0 = 0; kb0 < K; kb0 += 64) {
      __syncthreads();
      for (int e = tid; e < M * 8; e += 256) {
        int m = e >> 3, q = e & 7;
        *(v8h*)(wl + m * 72 + q * 8) =
            *(const v8h*)(W + (size_t)m * K + kb0 + q * 8);
      }
      stage_x(xl, 72, in, p0, P, K, 64, kb0, Lt, inL, in_t0, flags & 1);
      __syncthreads();
#pragma unroll
      for (int kb = 0; kb < 64; kb += 32) {
        {
          int tl = w;
          a0 = wmma32(load_a(wl, 72, (tl >> 2) * 16, kb),
                      load_b(xl, 72, (tl & 3) * 16, kb), a0);
        }
        if (tpw == 4) {
          int tl = w + 8;
          a1 = wmma32(load_a(wl, 72, (tl >> 2) * 16, kb),
                      load_b(xl, 72, (tl & 3) * 16, kb), a1);
          tl = w + 16;
          a2 = wmma32(load_a(wl, 72, (tl >> 2) * 16, kb),
                      load_b(xl, 72, (tl & 3) * 16, kb), a2);
          tl = w + 24;
          a3 = wmma32(load_a(wl, 72, (tl >> 2) * 16, kb),
                      load_b(xl, 72, (tl & 3) * 16, kb), a3);
        }
      }
    }
    store_tile(a0, w, p0, P, M, Lt, outL, out_t0, addL, add_t0, bias, add, out, flags);
    if (tpw == 4) {
      store_tile(a1, w + 8,  p0, P, M, Lt, outL, out_t0, addL, add_t0, bias, add, out, flags);
      store_tile(a2, w + 16, p0, P, M, Lt, outL, out_t0, addL, add_t0, bias, add, out, flags);
      store_tile(a3, w + 24, p0, P, M, Lt, outL, out_t0, addL, add_t0, bias, add, out, flags);
    }
  }
}

// ---------------------------------------------------------------------------
// maxpool k=3 s=2 (f32 out and f16 out variants)
// ---------------------------------------------------------------------------
__global__ void k_pool(const float* __restrict__ in, float* __restrict__ out,
                       int Lin, int Lout, int total) {
  int idx = blockIdx.x * blockDim.x + threadIdx.x;
  if (idx >= total) return;
  int t = idx % Lout;
  int nc = idx / Lout;
  const float* p = in + (size_t)nc * Lin + 2 * t;
  out[idx] = fmaxf(p[0], fmaxf(p[1], p[2]));
}

__global__ void k_pool_h(const float* __restrict__ in, _Float16* __restrict__ out,
                         int Lin, int Lout, int total) {
  int idx = blockIdx.x * blockDim.x + threadIdx.x;
  if (idx >= total) return;
  int t = idx % Lout;
  int nc = idx / Lout;
  const float* p = in + (size_t)nc * Lin + 2 * t;
  out[idx] = (_Float16)fmaxf(p[0], fmaxf(p[1], p[2]));
}

// ---------------------------------------------------------------------------
// FC GEMM (all-f16 I/O): out = relu(A[MbxK] @ W[NxK]^T + b), 64x64 tiles.
// Tiles streamed global->LDS by the Tensor Data Mover (wave 0 issues,
// s_wait_tensorcnt, then workgroup barrier publishes).
// ---------------------------------------------------------------------------
__global__ void __launch_bounds__(256)
k_fc(const _Float16* __restrict__ A, const _Float16* __restrict__ W,
     const float* __restrict__ bias, _Float16* __restrict__ out, int N, int K,
     int relu) {
  __shared__ __align__(16) _Float16 al[64 * 40];
  __shared__ __align__(16) _Float16 bl[64 * 40];
  int tid = threadIdx.x;
  int m0 = blockIdx.y * 64, n0 = blockIdx.x * 64;
  int w = tid >> 5, lane = tid & 31, half = lane >> 4, nn = lane & 15;
  int ti0 = w, ti1 = w + 8;                  // tile id = mi*4 + ni
  u32 al_off = (u32)(uintptr_t)&al[0];
  u32 bl_off = (u32)(uintptr_t)&bl[0];
  v8f acc0 = {}, acc1 = {};
  for (int k0 = 0; k0 < K; k0 += 32) {
    __syncthreads();
    if (w == 0) {
      tdm_load_2d_f16(al_off, A + (size_t)m0 * K + k0, (u32)K, 64u, (u32)K, 32u, 64u);
      tdm_load_2d_f16(bl_off, W + (size_t)n0 * K + k0, (u32)K, 64u, (u32)K, 32u, 64u);
      __builtin_amdgcn_s_wait_tensorcnt(0);
    }
    __syncthreads();
    acc0 = wmma32(load_a(al, 40, (ti0 >> 2) * 16, 0),
                  load_b(bl, 40, (ti0 & 3) * 16, 0), acc0);
    acc1 = wmma32(load_a(al, 40, (ti1 >> 2) * 16, 0),
                  load_b(bl, 40, (ti1 & 3) * 16, 0), acc1);
  }
#pragma unroll
  for (int v = 0; v < 8; ++v) {
    {
      int m = m0 + (ti0 >> 2) * 16 + half * 8 + v;
      int n = n0 + (ti0 & 3) * 16 + nn;
      float val = acc0[v] + bias[n];
      if (relu) val = fmaxf(val, 0.f);
      out[(size_t)m * N + n] = (_Float16)val;
    }
    {
      int m = m0 + (ti1 >> 2) * 16 + half * 8 + v;
      int n = n0 + (ti1 & 3) * 16 + nn;
      float val = acc1[v] + bias[n];
      if (relu) val = fmaxf(val, 0.f);
      out[(size_t)m * N + n] = (_Float16)val;
    }
  }
}

// ---------------------------------------------------------------------------
// head: logits = A @ W3^T + b3; softmax; argmax.
// d_out: [0,512) logits, [512,1024) probs, [1024,1280) argmax (as float)
// ---------------------------------------------------------------------------
__global__ void k_head(const _Float16* __restrict__ A, const float* __restrict__ W,
                       const float* __restrict__ b, float* __restrict__ out) {
  int bi = threadIdx.x;
  const _Float16* a = A + bi * 256;
  float l0 = b[0], l1 = b[1];
  for (int k = 0; k < 256; ++k) {
    float av = (float)a[k];
    l0 += av * W[k];
    l1 += av * W[256 + k];
  }
  out[bi * 2] = l0;
  out[bi * 2 + 1] = l1;
  float m = fmaxf(l0, l1);
  float e0 = __builtin_amdgcn_exp2f((l0 - m) * LOG2E);
  float e1 = __builtin_amdgcn_exp2f((l1 - m) * LOG2E);
  float s = __builtin_amdgcn_rcpf(e0 + e1);
  out[512 + bi * 2] = e0 * s;
  out[512 + bi * 2 + 1] = e1 * s;
  out[1024 + bi] = (l1 > l0) ? 1.f : 0.f;
}

// ---------------------------------------------------------------------------
extern "C" void kernel_launch(void* const* d_in, const int* in_sizes, int n_in,
                              void* d_out, int out_size, void* d_ws, size_t ws_size,
                              hipStream_t stream) {
  const float* x        = (const float*)d_in[0];
  const float* start_w  = (const float*)d_in[1];
  const float* filter_w = (const float*)d_in[2];
  const float* gate_w   = (const float*)d_in[3];
  const float* res_w    = (const float*)d_in[4];
  const float* skip_w   = (const float*)d_in[5];
  const float* end1_w   = (const float*)d_in[6];
  const float* end1_b   = (const float*)d_in[7];
  const float* end2_w   = (const float*)d_in[8];
  const float* end2_b   = (const float*)d_in[9];
  const float* fc_w1    = (const float*)d_in[10];
  const float* fc_b1    = (const float*)d_in[11];
  const float* fc_w2    = (const float*)d_in[12];
  const float* fc_b2    = (const float*)d_in[13];
  const float* fc_w3    = (const float*)d_in[14];
  const float* fc_b3    = (const float*)d_in[15];

  float* ws = (float*)d_ws;
  const size_t BUF = 10485760;               // 256*32*1280 floats
  float* B0 = ws;                            // residual (dilated layout)
  float* B1 = B0 + BUF;                      // gate output H
  float* B2 = B1 + BUF;                      // un-dilated skip source S
  float* B3 = B2 + BUF;                      // next residual HN
  const int LCAN = 1279;
  float* canvas = B3 + BUF;                  // (256,256,1279) tail-aligned skip
  float* y1 = canvas + (size_t)256 * 256 * LCAN;   // (256,128,1272)
  float* p1 = y1 + (size_t)256 * 128 * 1272;       // (256,128,635)
  float* y2 = p1 + (size_t)256 * 128 * 635;        // (256,32,635)
  // f16 workspace region
  _Float16* hbase = (_Float16*)(y2 + (size_t)256 * 32 * 635);
  _Float16* hWf  = hbase;                    // (6,32,64) remapped
  _Float16* hWg  = hWf + 12288;
  _Float16* hRes = hWg + 12288;              // (6,32,32)
  _Float16* hSkip= hRes + 6144;              // (6,256,32)
  _Float16* hE1  = hSkip + 49152;            // (128,256)
  _Float16* hE2  = hE1 + 32768;              // (32,128)
  _Float16* hFc1 = hE2 + 4096;               // (1024,10144)
  _Float16* hFc2 = hFc1 + 10387456;          // (256,1024)
  _Float16* hP2  = hFc2 + 262144;            // (256,10144) flattened pool2
  _Float16* hA1  = hP2 + 2596864;            // (256,1024)
  _Float16* hA2  = hA1 + 262144;             // (256,256)

  // ---- weight pre-conversion (f32 -> f16, once per launch) ----
  k_cvt_fg<<<(12288 + 255) / 256, 256, 0, stream>>>(filter_w, gate_w, hWf, hWg,
                                                    12288);
  k_cvt<<<(6144 + 255) / 256, 256, 0, stream>>>(res_w, hRes, 6144);
  k_cvt<<<(49152 + 255) / 256, 256, 0, stream>>>(skip_w, hSkip, 49152);
  k_cvt<<<(32768 + 255) / 256, 256, 0, stream>>>(end1_w, hE1, 32768);
  k_cvt<<<(4096 + 255) / 256, 256, 0, stream>>>(end2_w, hE2, 4096);
  k_cvt<<<(10387456 + 255) / 256, 256, 0, stream>>>(fc_w1, hFc1, 10387456);
  k_cvt<<<(262144 + 255) / 256, 256, 0, stream>>>(fc_w2, hFc2, 262144);

  // start conv: x -> B0 (256,32,1280)
  {
    int total = 256 * 32 * 1280;
    k_start<<<(total + 255) / 256, 256, 0, stream>>>(x, start_w, B0, total);
  }

  // layer:        0     1     2     3     4     5
  const int Ns[6]   = {256, 512, 1024, 256, 512, 1024};
  const int Lrs[6]  = {1280, 640, 320, 1276, 638, 319};
  const int ds[6]   = {1, 2, 4, 1, 2, 4};

  for (int i = 0; i < 6; ++i) {
    int N = Ns[i], Lr = Lrs[i], Lo = Lr - 1, d = ds[i];

    if (i == 1 || i == 2 || i == 4 || i == 5) {     // fold with f=2
      int Nin = N / 2;
      int Lin = Lrs[i - 1] - 1;
      int pad = 2 * Lr - Lin;                       // == 1
      int total = N * 32 * Lr;
      k_fold<<<(total + 255) / 256, 256, 0, stream>>>(B3, B0, Nin, Lin, 2, Lr,
                                                      pad, total);
    } else if (i == 3) {                            // unfold with g=4
      int total = 256 * 32 * 1276;
      k_unfold<<<(total + 255) / 256, 256, 0, stream>>>(B3, B0, 256, 319, 4,
                                                        total);
    }

    int P = N * Lo;
    k_conv2gate<<<(P + 63) / 64, 256, 0, stream>>>(
        B0, hWf + i * 2048, hWg + i * 2048, B1, Lr, P);

    float* S = B1;
    int Ls = Lo;
    if (d > 1) {
      Ls = Lo * d;
      int total = 256 * 32 * Ls;
      k_unfold<<<(total + 255) / 256, 256, 0, stream>>>(B1, B2, 256, Lo, d,
                                                        total);
      S = B2;
    }

    {
      int Pp = 256 * Ls;
      int flags = (i == 0) ? 0 : 4;
      k_conv1x1<<<(Pp + 63) / 64, 256, 0, stream>>>(
          S, hSkip + i * 256 * 32, nullptr, nullptr, canvas, 256, 32, Ls, Ls, 0,
          LCAN, LCAN - Ls, 0, 0, Pp, flags);
    }

    if (i < 5) {
      k_conv1x1<<<(P + 63) / 64, 256, 0, stream>>>(
          B1, hRes + i * 1024, nullptr, B0, B3, 32, 32, Lo, Lo, 0, Lo, 0, Lr, 1,
          P, 0);
    }
  }

  // end head
  {
    int P = 256 * 1272;
    k_conv1x1<<<(P + 63) / 64, 256, 0, stream>>>(
        canvas, hE1, end1_b, nullptr, y1, 128, 256, 1272, LCAN, LCAN - 1272,
        1272, 0, 0, 0, P, 1 | 2);
  }
  {
    int total = 256 * 128 * 635;
    k_pool<<<(total + 255) / 256, 256, 0, stream>>>(y1, p1, 1272, 635, total);
  }
  {
    int P = 256 * 635;
    k_conv1x1<<<(P + 63) / 64, 256, 0, stream>>>(
        p1, hE2, end2_b, nullptr, y2, 32, 128, 635, 635, 0, 635, 0, 0, 0, P, 2);
  }
  {
    int total = 256 * 32 * 317;
    k_pool_h<<<(total + 255) / 256, 256, 0, stream>>>(y2, hP2, 635, 317, total);
  }

  // MLP with TDM-fed GEMMs
  {
    dim3 g(1024 / 64, 256 / 64);
    k_fc<<<g, 256, 0, stream>>>(hP2, hFc1, fc_b1, hA1, 1024, 10144, 1);
  }
  {
    dim3 g(256 / 64, 256 / 64);
    k_fc<<<g, 256, 0, stream>>>(hA1, hFc2, fc_b2, hA2, 256, 1024, 1);
  }
  k_head<<<1, 256, 0, stream>>>(hA2, fc_w3, fc_b3, (float*)d_out);
}
